// AttnBlock_67044439490881
// MI455X (gfx1250) — matrix operand order repaired and checked by reading, not compiled, not run on hardware
//
#include <hip/hip_runtime.h>
#include <hip/hip_bf16.h>

typedef _Float16 v16h __attribute__((ext_vector_type(16)));
typedef _Float16 v8h  __attribute__((ext_vector_type(8)));
typedef float    v8f  __attribute__((ext_vector_type(8)));

#define DIMQ 256
#define DIMV 384
#define NH   6
#define DH   64
#define NN   1024
#define NB   8
#define LSTR 1032           // 16-row logits strip stride (pad 8 dwords: conflict-free)
#define SCALE 0.051031036307982884f   // 1/sqrt(384)

// ---- wave32 reductions -------------------------------------------------
__device__ __forceinline__ float wave_sum(float v) {
#pragma unroll
  for (int o = 16; o > 0; o >>= 1) v += __shfl_xor(v, o, 32);
  return v;
}
__device__ __forceinline__ float wave_max(float v) {
#pragma unroll
  for (int o = 16; o > 0; o >>= 1) v = fmaxf(v, __shfl_xor(v, o, 32));
  return v;
}

// A-fragment (16x32 f16): lane(hi) holds K = hi*16..+15 (two 16B groups)
__device__ __forceinline__ v16h load_a_frag(const _Float16* rowK0, int hi) {
  v8h g0 = *(const v8h*)(rowK0 + hi * 16);
  v8h g1 = *(const v8h*)(rowK0 + hi * 16 + 8);
  return __builtin_shufflevector(g0, g1, 0,1,2,3,4,5,6,7,8,9,10,11,12,13,14,15);
}

// ---- kernel 0: stage WQ/WK as f16, B-fragment friendly (K-major per column)
__global__ void __launch_bounds__(256)
prep_w(const float* __restrict__ WQ, const float* __restrict__ WK,
       _Float16* __restrict__ WQt, _Float16* __restrict__ WKt) {
  int t = blockIdx.x * blockDim.x + threadIdx.x;     // 2 * 8 * 384 tasks
  if (t >= 2 * 8 * DIMV) return;
  int mat = t / (8 * DIMV);
  int r   = t - mat * (8 * DIMV);
  int kc  = r / DIMV;
  int col = r - kc * DIMV;
  const float* src = mat ? WK : WQ;
  _Float16*    dst = mat ? WKt : WQt;
#pragma unroll
  for (int i = 0; i < 32; ++i)
    dst[((size_t)kc * DIMV + col) * 32 + i] =
        (_Float16)src[(size_t)(kc * 32 + i) * DIMV + col];
}

// ---- kernel 1: X[8192x256] @ W[256x384] + b -> f16 [b,h,n,64]
__global__ void __launch_bounds__(128)
proj(const float* __restrict__ X, const _Float16* __restrict__ Wt,
     const float* __restrict__ bias, _Float16* __restrict__ outp) {
  const int lane = threadIdx.x & 31;
  const int wv   = threadIdx.x >> 5;
  const int nlo  = lane & 15;
  const int hi   = lane >> 4;
  const int m0   = blockIdx.x * 16;                 // global row tile (8192 rows)

  v8f acc[6];
#pragma unroll
  for (int t = 0; t < 6; ++t) acc[t] = (v8f)(0.0f);

#pragma unroll
  for (int kc = 0; kc < 8; ++kc) {
    // A fragment from f32 rows: two contiguous 32B groups, convert to f16
    const float* abase = X + (size_t)(m0 + nlo) * DIMQ + kc * 32;
    v8f q0 = *(const v8f*)(abase + hi * 16);
    v8f q1 = *(const v8f*)(abase + hi * 16 + 8);
    v16h afrag;
#pragma unroll
    for (int i = 0; i < 8; ++i) { afrag[i] = (_Float16)q0[i]; afrag[8 + i] = (_Float16)q1[i]; }
#pragma unroll
    for (int t = 0; t < 6; ++t) {
      int col = wv * 96 + t * 16 + nlo;             // B lane = output column
      v16h bfrag = *(const v16h*)(Wt + ((size_t)kc * DIMV + col) * 32 + hi * 16);
      acc[t] = __builtin_amdgcn_wmma_f32_16x16x32_f16(
          false, afrag, false, bfrag, (short)0, acc[t], false, false);
    }
  }
  // epilogue: + bias, convert, scatter to [b,h,n,64]
#pragma unroll
  for (int t = 0; t < 6; ++t) {
    int col = wv * 96 + t * 16 + nlo;
    float bv = bias[col];
    int h  = col >> 6;
    int dc = col & 63;
#pragma unroll
    for (int r = 0; r < 8; ++r) {
      int row = m0 + r + hi * 8;
      int bb  = row >> 10;
      int nn  = row & 1023;
      outp[(((size_t)bb * NH + h) * NN + nn) * DH + dc] = (_Float16)(acc[t][r] + bv);
    }
  }
}

// ---- kernel 2: QK^T (WMMA) -> masked sparsemax -> head-major store
__global__ void __launch_bounds__(128)
attn(const _Float16* __restrict__ qbuf, const _Float16* __restrict__ kbuf,
     const float* __restrict__ Aadj, float* __restrict__ outp) {
  extern __shared__ float lds[];                    // 16 * LSTR floats (66 KB)
  const int lane = threadIdx.x & 31;
  const int wv   = threadIdx.x >> 5;
  const int nlo  = lane & 15;
  const int hi   = lane >> 4;
  const int blk  = blockIdx.x;
  const int bh   = blk >> 6;
  const int mt   = blk & 63;
  const int b    = bh / NH;
  const int h    = bh - b * NH;
  const int m0   = mt * 16;

  const _Float16* qbase = qbuf + ((size_t)bh * NN + m0) * DH;
  const _Float16* kbase = kbuf + (size_t)bh * NN * DH;

  // prefetch this wave's adjacency rows for the sparsemax phase
#pragma unroll
  for (int rr = 0; rr < 4; ++rr)
    __builtin_prefetch(Aadj + ((size_t)b * NN + m0 + wv * 4 + rr) * NN, 0, 0);

  // persistent A fragments: q rows m0..m0+15, K chunks {0,1} of d=64
  const _Float16* qrow = qbase + (size_t)nlo * DH;
  v16h a0 = load_a_frag(qrow, hi);
  v16h a1 = load_a_frag(qrow + 32, hi);

  v8f acc[16];
#pragma unroll
  for (int jt = 0; jt < 16; ++jt) acc[jt] = (v8f)(0.0f);

#pragma unroll
  for (int jt = 0; jt < 16; ++jt) {
    int j = wv * 256 + jt * 16 + nlo;               // B lane = k row (column of K^T)
    const _Float16* krow = kbase + (size_t)j * DH;
    v16h b0 = *(const v16h*)(krow + hi * 16);       // B: K contiguous per lane
    v16h b1 = *(const v16h*)(krow + 32 + hi * 16);
    acc[jt] = __builtin_amdgcn_wmma_f32_16x16x32_f16(
        false, a0, false, b0, (short)0, acc[jt], false, false);
    acc[jt] = __builtin_amdgcn_wmma_f32_16x16x32_f16(
        false, a1, false, b1, (short)0, acc[jt], false, false);
  }

  // stage scaled logits strip to LDS (lanes 0-15 / 16-31 hit disjoint banks)
#pragma unroll
  for (int jt = 0; jt < 16; ++jt) {
#pragma unroll
    for (int r = 0; r < 8; ++r)
      lds[(size_t)(r + hi * 8) * LSTR + wv * 256 + jt * 16 + nlo] = acc[jt][r] * SCALE;
  }
  __syncthreads();

  // sparsemax: each wave owns 4 full rows (32 logits per lane, kept in VGPRs)
#pragma unroll 1
  for (int rr = 0; rr < 4; ++rr) {
    const int row = wv * 4 + rr;
    const float* arow = Aadj + ((size_t)b * NN + m0 + row) * NN;
    float z[32];
    float mx = -3.0e38f;
#pragma unroll
    for (int c = 0; c < 32; ++c) {
      float zv = lds[(size_t)row * LSTR + c * 32 + lane];
      float av = arow[c * 32 + lane];
      zv = (av != 0.0f) ? zv : -1.0e30f;            // hard mask -> never in support
      z[c] = zv;
      mx = fmaxf(mx, zv);
    }
    mx = wave_max(mx);                               // self-loop guarantees finite
    // tau in [mx-1, mx): bisect monotone f(tau) = sum max(z-tau,0) against 1
    float lo = mx - 1.0f, hh = mx;
#pragma unroll 1
    for (int it = 0; it < 24; ++it) {
      float mid = 0.5f * (lo + hh);
      float s = 0.0f;
#pragma unroll
      for (int c = 0; c < 32; ++c) s += fmaxf(z[c] - mid, 0.0f);
      s = wave_sum(s);
      if (s >= 1.0f) lo = mid; else hh = mid;
    }
    // exact recovery: support = {z > lo}; tau = (sum - 1)/k
    float s = 0.0f, cnt = 0.0f;
#pragma unroll
    for (int c = 0; c < 32; ++c) {
      if (z[c] > lo) { s += z[c]; cnt += 1.0f; }
    }
    s = wave_sum(s); cnt = wave_sum(cnt);
    float tau = (s - 1.0f) / cnt;
    // head-major output: out[b][m][h*1024 + j]
    float* orow = outp + ((size_t)(b * NN + m0 + row)) * (NH * NN) + (size_t)h * NN;
#pragma unroll
    for (int c = 0; c < 32; ++c)
      orow[c * 32 + lane] = fmaxf(z[c] - tau, 0.0f);
  }
}

extern "C" void kernel_launch(void* const* d_in, const int* in_sizes, int n_in,
                              void* d_out, int out_size, void* d_ws, size_t ws_size,
                              hipStream_t stream) {
  (void)in_sizes; (void)n_in; (void)out_size; (void)ws_size;
  const float* Q  = (const float*)d_in[0];
  const float* V  = (const float*)d_in[1];
  const float* A  = (const float*)d_in[2];
  const float* WQ = (const float*)d_in[3];
  const float* bQ = (const float*)d_in[4];
  const float* WK = (const float*)d_in[5];
  const float* bK = (const float*)d_in[6];
  float* out = (float*)d_out;

  char* ws = (char*)d_ws;
  _Float16* WQt = (_Float16*)ws;                          // 8*384*32 f16 = 192 KB
  _Float16* WKt = (_Float16*)(ws + 196608);               // 192 KB
  _Float16* qb  = (_Float16*)(ws + 393216);               // 48*1024*64 f16 = 6 MB
  _Float16* kb  = (_Float16*)(ws + 393216 + 6291456);     // 6 MB  (total ~12.4 MB)

  prep_w<<<24, 256, 0, stream>>>(WQ, WK, WQt, WKt);
  proj<<<512, 128, 0, stream>>>(Q, WQt, bQ, qb);
  proj<<<512, 128, 0, stream>>>(V, WKt, bK, kb);
  attn<<<NB * NH * (NN / 16), 128, 16 * LSTR * sizeof(float), stream>>>(qb, kb, A, out);
}
